// Neurons_LIF_20117626814651
// MI455X (gfx1250) — matrix operand order repaired and checked by reading, not compiled, not run on hardware
//
#include <hip/hip_runtime.h>
#include <math.h>

// ---------------- problem constants ----------------
#define NN 2048   // recurrent neurons (K dim, also r_out N dim)
#define EN 1638   // excitatory count
#define ON 512    // readout size
#define BB 4096   // batch (M dim)

typedef __bf16 bf16;
typedef bf16  v16bf __attribute__((ext_vector_type(16)));
typedef bf16  v8bf  __attribute__((ext_vector_type(8)));
typedef float v8f   __attribute__((ext_vector_type(8)));

// ---------------- helpers ----------------
static __device__ __forceinline__ v16bf cat8(v8bf a, v8bf b) {
  return __builtin_shufflevector(a, b, 0,1,2,3,4,5,6,7,8,9,10,11,12,13,14,15);
}

static __device__ __forceinline__ void split_bf(float v, bf16& h, bf16& l) {
  h = (bf16)v;                 // RNE truncation to bf16 (high part)
  l = (bf16)(v - (float)h);    // residual (low part), |l| <= 2^-8 |v|
}

// ---------------- LIF elementwise: u = relu(0.9*x + 0.1*(i_+b)) ----------------
__global__ __launch_bounds__(256)
void lif_u_kernel(const float* __restrict__ iin, const float* __restrict__ x,
                  const float* __restrict__ b, float* __restrict__ u) {
  size_t idx4 = (size_t)blockIdx.x * blockDim.x + threadIdx.x;  // one float4 per thread
  size_t base = idx4 * 4;
  int col = (int)(base & (NN - 1));            // NN is a power of two
  float4 iv = *(const float4*)(iin + base);
  float4 xv = *(const float4*)(x + base);
  float4 bv = *(const float4*)(b + col);
  float4 o;
  o.x = fmaxf(0.0f, 0.9f * xv.x + 0.1f * (iv.x + bv.x));
  o.y = fmaxf(0.0f, 0.9f * xv.y + 0.1f * (iv.y + bv.y));
  o.z = fmaxf(0.0f, 0.9f * xv.z + 0.1f * (iv.z + bv.z));
  o.w = fmaxf(0.0f, 0.9f * xv.w + 0.1f * (iv.w + bv.w));
  *(float4*)(u + base) = o;
}

// ---------------- split-bf16 WMMA GEMM: C = A @ eff(W) ----------------
// A: [M x K] fp32 row-major (u).  W: [K x Nc] fp32 row-major (raw weights).
// eff(W)[k][n] = sign(k) * |W[k][n]|, sign(k) = +1 if k < EN else -1,
// and (diagMask) eff zeroed when k == n (no_self for the recurrent matrix).
// Block: 256 threads = 8 wave32 waves; tile 128(M) x 128(N); K staged by 32.
// Per wave: 64x32 region = 4x2 accumulators of v_wmma_f32_16x16x32_bf16,
// 3 WMMA per K-chunk per accumulator (hi*hi + hi*lo + lo*hi).
__global__ __launch_bounds__(256)
void gemm_split_bf16(const float* __restrict__ A, const float* __restrict__ W,
                     float* __restrict__ C, int M, int Nc, int K, int diagMask) {
  // padded to 40 bf16 (80B): keeps 16B alignment, conflict-free fragment reads
  __shared__ bf16 sAh[128][40];
  __shared__ bf16 sAl[128][40];
  __shared__ bf16 sBh[128][40];   // B stored transposed: [n][k]
  __shared__ bf16 sBl[128][40];

  const int tid  = threadIdx.x;
  const int m0   = blockIdx.x * 128;
  const int n0   = blockIdx.y * 128;

  const int lane = tid & 31;
  const int wave = tid >> 5;
  const int wm   = wave & 1;      // which 64-row half
  const int wn   = wave >> 1;     // which 32-col quarter
  const int half = lane >> 4;     // lane half (maps K-chunks per ISA layout)
  const int l15  = lane & 15;

  v8f acc[4][2];
#pragma unroll
  for (int i = 0; i < 4; ++i)
#pragma unroll
    for (int j = 0; j < 2; ++j) acc[i][j] = (v8f)0.0f;

  // loader decomposition
  const int arow = tid >> 3;      // 0..31 (A row within pass)
  const int akq  = tid & 7;       // float4 index along K
  const int bnq  = tid & 31;      // float4 index along N
  const int bk   = tid >> 5;      // 0..7 (B k-row within pass)

  for (int k0 = 0; k0 < K; k0 += 32) {
    // ---- global -> LDS, A tile 128x32 with on-the-fly hi/lo split ----
#pragma unroll
    for (int rr = 0; rr < 128; rr += 32) {
      const int row = arow + rr;
      float4 v = *(const float4*)(&A[(size_t)(m0 + row) * K + k0 + akq * 4]);
      union { bf16 e[4]; uint2 u2; } ph, pl;
      split_bf(v.x, ph.e[0], pl.e[0]);
      split_bf(v.y, ph.e[1], pl.e[1]);
      split_bf(v.z, ph.e[2], pl.e[2]);
      split_bf(v.w, ph.e[3], pl.e[3]);
      *(uint2*)(&sAh[row][akq * 4]) = ph.u2;
      *(uint2*)(&sAl[row][akq * 4]) = pl.u2;
    }
    // ---- global -> LDS, B tile 32x128, transposed + Dale sign/abs/diag ----
#pragma unroll
    for (int kk = 0; kk < 32; kk += 8) {
      const int gk  = k0 + kk + bk;
      const float s = (gk < EN) ? 1.0f : -1.0f;
      float4 v = *(const float4*)(&W[(size_t)gk * Nc + n0 + bnq * 4]);
      float w[4] = {v.x, v.y, v.z, v.w};
#pragma unroll
      for (int j = 0; j < 4; ++j) {
        const int n = bnq * 4 + j;
        float e = s * fabsf(w[j]);
        if (diagMask && (gk == n0 + n)) e = 0.0f;
        bf16 h, l;
        split_bf(e, h, l);
        sBh[n][kk + bk] = h;
        sBl[n][kk + bk] = l;
      }
    }
    __syncthreads();

    // prefetch next K-tile while this stage computes
    if (k0 + 32 < K) {
      __builtin_prefetch(&A[(size_t)(m0 + arow) * K + (k0 + 32) + akq * 4], 0, 1);
      __builtin_prefetch(&W[(size_t)(k0 + 32 + bk) * Nc + n0 + bnq * 4], 0, 1);
    }

    // ---- B fragments (16-bit B 32x16: lanes<16 K {0..7,16..23}; lanes>=16 {8..15,24..31}) ----
    v16bf bh[2], bl[2];
#pragma unroll
    for (int nf = 0; nf < 2; ++nf) {
      const int ncol = wn * 32 + nf * 16 + l15;
      const v8bf* ph = (const v8bf*)(&sBh[ncol][half * 8]);
      const v8bf* pl = (const v8bf*)(&sBl[ncol][half * 8]);
      bh[nf] = cat8(ph[0], ph[2]);
      bl[nf] = cat8(pl[0], pl[2]);
    }

    // ---- A fragments + 3-term split WMMA ----
#pragma unroll
    for (int mf = 0; mf < 4; ++mf) {
      const int rrow = wm * 64 + mf * 16 + l15;
      // ISA 16-bit A 16x32: lanes<16 hold K {0..7,16..23}; lanes>=16 {8..15,24..31}
      v8bf a0h = *(const v8bf*)(&sAh[rrow][half * 8]);
      v8bf a1h = *(const v8bf*)(&sAh[rrow][16 + half * 8]);
      v8bf a0l = *(const v8bf*)(&sAl[rrow][half * 8]);
      v8bf a1l = *(const v8bf*)(&sAl[rrow][16 + half * 8]);
      v16bf ah = cat8(a0h, a1h);
      v16bf al = cat8(a0l, a1l);
#pragma unroll
      for (int nf = 0; nf < 2; ++nf) {
        acc[mf][nf] = __builtin_amdgcn_wmma_f32_16x16x32_bf16(
            false, ah, false, bh[nf], (short)0, acc[mf][nf], false, false);
        acc[mf][nf] = __builtin_amdgcn_wmma_f32_16x16x32_bf16(
            false, ah, false, bl[nf], (short)0, acc[mf][nf], false, false);
        acc[mf][nf] = __builtin_amdgcn_wmma_f32_16x16x32_bf16(
            false, al, false, bh[nf], (short)0, acc[mf][nf], false, false);
      }
    }
    __syncthreads();
  }

  // ---- epilogue: C/D 16x16 f32 layout: VGPR p -> M=p (lanes<16) / M=p+8 ----
#pragma unroll
  for (int mf = 0; mf < 4; ++mf) {
#pragma unroll
    for (int nf = 0; nf < 2; ++nf) {
      const int row = m0 + wm * 64 + mf * 16 + half * 8;
      const int col = n0 + wn * 32 + nf * 16 + l15;
#pragma unroll
      for (int p = 0; p < 8; ++p) {
        C[(size_t)(row + p) * Nc + col] = acc[mf][nf][p];
      }
    }
  }
}

// ---------------- launcher ----------------
extern "C" void kernel_launch(void* const* d_in, const int* in_sizes, int n_in,
                              void* d_out, int out_size, void* d_ws, size_t ws_size,
                              hipStream_t stream) {
  (void)in_sizes; (void)n_in; (void)out_size; (void)d_ws; (void)ws_size;
  const float* i_ = (const float*)d_in[0];   // [BB, NN]
  const float* x  = (const float*)d_in[1];   // [BB, NN]
  const float* r  = (const float*)d_in[2];   // [NN, NN]
  const float* f  = (const float*)d_in[3];   // [NN, ON]
  const float* b  = (const float*)d_in[4];   // [NN]

  float* f_out = (float*)d_out;                      // [BB, ON]
  float* r_out = f_out + (size_t)BB * ON;            // [BB, NN]
  float* u     = r_out + (size_t)BB * NN;            // [BB, NN]  (also GEMM A operand)

  // 1) LIF state update + relu -> u (written into its output slot, reused as A)
  lif_u_kernel<<<(BB * NN / 4) / 256, 256, 0, stream>>>(i_, x, b, u);

  // 2) f_out = u @ (sign*|f|)
  gemm_split_bf16<<<dim3(BB / 128, ON / 128), 256, 0, stream>>>(
      u, f, f_out, BB, ON, NN, /*diagMask=*/0);

  // 3) r_out = u @ (sign*|r|, zero diagonal)
  gemm_split_bf16<<<dim3(BB / 128, NN / 128), 256, 0, stream>>>(
      u, r, r_out, BB, NN, NN, /*diagMask=*/1);
}
